// MultiHeadSelfAttention_38981123178826
// MI455X (gfx1250) — matrix operand order repaired and checked by reading, not compiled, not run on hardware
//
#include <hip/hip_runtime.h>
#include <hip/hip_bf16.h>

typedef __bf16 bf16_t;
typedef __attribute__((ext_vector_type(16))) __bf16 v16bf;
typedef __attribute__((ext_vector_type(8)))  __bf16 v8bf;
typedef __attribute__((ext_vector_type(8)))  float  v8f;
typedef __attribute__((ext_vector_type(4)))  float  v4f;

#define BB 8
#define CC 256
#define NN 4096
#define CQv 32
#define OC_TOT 320   // 32 (q) + 32 (k) + 256 (v) output channels

// workspace layout (bytes): W bf16 160KB | Q 2MB | K 2MB | V 16MB  => needs ~21MB
#define WS_W 0u
#define WS_Q (1u << 20)
#define WS_K (3u << 20)
#define WS_V (5u << 20)

__device__ __forceinline__ v16bf cat16(v8bf lo, v8bf hi) {
    return __builtin_shufflevector(lo, hi, 0,1,2,3,4,5,6,7,8,9,10,11,12,13,14,15);
}

// A-operand (16x32 bf16, M x K) from row-major storage, tile origin at `base`.
// ISA layout: lane m<16 holds K{0..7,16..23}, lane m+16 holds K{8..15,24..31}.
__device__ __forceinline__ v16bf load_A_rm(const bf16_t* base, int rowStride, int lane) {
    int r = lane & 15, half = lane >> 4;
    const bf16_t* p = base + r * rowStride + half * 8;
    v8bf lo = *(const v8bf*)p;          // K = half*8 .. half*8+7
    v8bf hi = *(const v8bf*)(p + 16);   // K = 16+half*8 ..
    return cat16(lo, hi);
}

// B-operand (32x16 bf16, K x N) where each operand COLUMN is contiguous in
// memory (stride `colStride` between columns). ISA layout: lanes 0-15 hold
// K{0..7,16..23} of column n, lanes 16-31 hold K{8..15,24..31}.
__device__ __forceinline__ v16bf load_B_col(const bf16_t* base, int colStride, int lane) {
    int c = lane & 15, koff = (lane >> 4) * 8;
    const bf16_t* p = base + c * colStride + koff;
    v8bf lo = *(const v8bf*)p;          // K = koff .. koff+7
    v8bf hi = *(const v8bf*)(p + 16);   // K = 16+koff .. 16+koff+7
    return cat16(lo, hi);
}

// ---------------- kernel 0: fp32 weights -> concatenated bf16 [320][256] ----
__global__ void wconv_kernel(const float* __restrict__ wq, const float* __restrict__ wk,
                             const float* __restrict__ wv, bf16_t* __restrict__ W) {
    int row = blockIdx.x;
    int t = threadIdx.x;
    const float* src = (row < 32) ? (wq + row * CC)
                     : (row < 64) ? (wk + (row - 32) * CC)
                                  : (wv + (row - 64) * CC);
    W[row * CC + t] = (bf16_t)src[t];
}

// ---------------- kernel 1: fused QKV projection (W[320x256] @ X[256xN]) ----
__global__ __launch_bounds__(128) void qkv_proj_kernel(
        const float* __restrict__ x,
        const float* __restrict__ bq, const float* __restrict__ bk, const float* __restrict__ bv,
        const bf16_t* __restrict__ W,
        bf16_t* __restrict__ Q, bf16_t* __restrict__ Kp, bf16_t* __restrict__ V) {
    // xT in LDS: [64 n][264 c] bf16 (stride 264 -> 528B rows, 16B aligned,
    // 132-dword bank stride => conflict-free 16-lane column reads)
    __shared__ bf16_t xT[64 * 264];
    int b  = blockIdx.y;
    int n0 = blockIdx.x * 64;
    int t  = threadIdx.x;
    const float* xb = x + (size_t)b * CC * NN;

    // cooperative load + transpose + bf16 convert of the [256 c][64 n] x tile
    for (int i = 0; i < 32; ++i) {
        int id = t + 128 * i;            // 4096 float4's
        int c  = id >> 4;
        int nn = (id & 15) * 4;
        v4f xv = *(const v4f*)(xb + (size_t)c * NN + n0 + nn);
#pragma unroll
        for (int j = 0; j < 4; ++j) xT[(nn + j) * 264 + c] = (bf16_t)xv[j];
    }
    __syncthreads();

    int wave = t >> 5, lane = t & 31;
    int col = lane & 15, half = lane >> 4;

    for (int rt = wave * 5; rt < wave * 5 + 5; ++rt) {   // 20 row-tiles / 4 waves
        v8f acc[4] = {};                                  // 4 n-tiles of 16
        for (int k0 = 0; k0 < CC; k0 += 32) {
            v16bf a = load_A_rm(W + rt * 16 * CC + k0, CC, lane);
#pragma unroll
            for (int nt = 0; nt < 4; ++nt) {
                v16bf bm = load_B_col(&xT[(nt * 16) * 264 + k0], 264, lane);
                acc[nt] = __builtin_amdgcn_wmma_f32_16x16x32_bf16(
                    false, a, false, bm, (short)0, acc[nt], false, false);
            }
        }
        // epilogue: + bias, scatter to Q [N][32], K [N][32], V [C][N] (bf16)
#pragma unroll
        for (int nt = 0; nt < 4; ++nt) {
#pragma unroll
            for (int r = 0; r < 8; ++r) {
                int oc = rt * 16 + r + half * 8;
                int n  = n0 + nt * 16 + col;
                float bias = (oc < 32) ? bq[oc] : (oc < 64 ? bk[oc - 32] : bv[oc - 64]);
                bf16_t val = (bf16_t)(acc[nt][r] + bias);
                if (oc < 32)      Q [((size_t)b * NN + n) * CQv + oc]        = val;
                else if (oc < 64) Kp[((size_t)b * NN + n) * CQv + (oc - 32)] = val;
                else              V [((size_t)b * CC + (oc - 64)) * NN + n]  = val;
            }
        }
    }
}

// ---------------- kernel 2: fused flash attention + residual ---------------
// One wave owns 16 query rows x 128 value channels (64 accumulator VGPRs).
// Wave pairs share a query tile; each covers one half of C.
__global__ __launch_bounds__(128) void attn_kernel(
        const float* __restrict__ x, const float* __restrict__ gamma,
        const bf16_t* __restrict__ Q, const bf16_t* __restrict__ Kp,
        const bf16_t* __restrict__ V, float* __restrict__ out) {
    __shared__ bf16_t Plds[4][16 * 40];   // per-wave P staging, 80B row stride
    int b    = blockIdx.y;
    int t    = threadIdx.x;
    int wave = t >> 5, lane = t & 31;
    int n0    = blockIdx.x * 32 + (wave >> 1) * 16;  // query tile
    int chalf = wave & 1;                            // channel half (0..127 / 128..255)
    int col = lane & 15, half = lane >> 4;

    const bf16_t* Qb = Q  + (size_t)b * NN * CQv;
    const bf16_t* Kb = Kp + (size_t)b * NN * CQv;
    const bf16_t* Vb = V  + (size_t)b * CC * NN;
    bf16_t* P = &Plds[wave][0];

    // Q A-operand: one 16x32 tile covers the whole Cq=32 contraction
    v16bf qa = load_A_rm(Qb + (size_t)n0 * CQv, CQv, lane);

    v8f acc[8] = {};
    float mrow[8], lrow[8];
#pragma unroll
    for (int r = 0; r < 8; ++r) { mrow[r] = -1e30f; lrow[r] = 0.f; }

    for (int j0 = 0; j0 < NN; j0 += 32) {
        __builtin_prefetch(Kb + (size_t)(j0 + 32) * CQv, 0, 0);
        v16bf kb0 = load_B_col(Kb + (size_t)(j0)      * CQv, CQv, lane);
        v16bf kb1 = load_B_col(Kb + (size_t)(j0 + 16) * CQv, CQv, lane);
        v8f z = {};
        v8f s0 = __builtin_amdgcn_wmma_f32_16x16x32_bf16(false, qa, false, kb0, (short)0, z, false, false);
        v8f s1 = __builtin_amdgcn_wmma_f32_16x16x32_bf16(false, qa, false, kb1, (short)0, z, false, false);

        // online softmax: row r lives in VGPR slot r across a 16-lane group
#pragma unroll
        for (int r = 0; r < 8; ++r) {
            float tmax = fmaxf(s0[r], s1[r]);
            tmax = fmaxf(tmax, __shfl_xor(tmax, 1, 32));
            tmax = fmaxf(tmax, __shfl_xor(tmax, 2, 32));
            tmax = fmaxf(tmax, __shfl_xor(tmax, 4, 32));
            tmax = fmaxf(tmax, __shfl_xor(tmax, 8, 32));
            float mnew  = fmaxf(mrow[r], tmax);
            float scale = __expf(mrow[r] - mnew);
            float p0 = __expf(s0[r] - mnew);
            float p1 = __expf(s1[r] - mnew);
            float ps = p0 + p1;
            ps += __shfl_xor(ps, 1, 32);
            ps += __shfl_xor(ps, 2, 32);
            ps += __shfl_xor(ps, 4, 32);
            ps += __shfl_xor(ps, 8, 32);
            lrow[r] = lrow[r] * scale + ps;
            mrow[r] = mnew;
#pragma unroll
            for (int ct = 0; ct < 8; ++ct) acc[ct][r] *= scale;
            int row = r + half * 8;
            P[row * 40 + col]      = (bf16_t)p0;
            P[row * 40 + 16 + col] = (bf16_t)p1;
        }
        asm volatile("s_wait_dscnt 0" ::: "memory");  // intra-wave LDS RAW

        v16bf pa = load_A_rm(P, 40, lane);            // P as 16x32 A-operand
#pragma unroll
        for (int ct = 0; ct < 8; ++ct) {
            const bf16_t* vtile = Vb + (size_t)(chalf * 128 + ct * 16) * NN + j0;
            v16bf vb = load_B_col(vtile, NN, lane);   // V^T tile: columns contiguous
            acc[ct] = __builtin_amdgcn_wmma_f32_16x16x32_bf16(
                false, pa, false, vb, (short)0, acc[ct], false, false);
        }
    }

    // epilogue: normalize, residual, write out[b][c][n]
    float g = gamma[0];
    const float* xb = x   + (size_t)b * CC * NN;
    float*       ob = out + (size_t)b * CC * NN;
#pragma unroll
    for (int ct = 0; ct < 8; ++ct) {
#pragma unroll
        for (int r = 0; r < 8; ++r) {
            int c = chalf * 128 + ct * 16 + col;
            int n = n0 + r + half * 8;
            size_t idx = (size_t)c * NN + n;
            float val = acc[ct][r] / lrow[r];
            ob[idx] = g * val + xb[idx];
        }
    }
}

extern "C" void kernel_launch(void* const* d_in, const int* in_sizes, int n_in,
                              void* d_out, int out_size, void* d_ws, size_t ws_size,
                              hipStream_t stream) {
    const float* x     = (const float*)d_in[0];
    const float* wq    = (const float*)d_in[1];
    const float* bq    = (const float*)d_in[2];
    const float* wk    = (const float*)d_in[3];
    const float* bk    = (const float*)d_in[4];
    const float* wv    = (const float*)d_in[5];
    const float* bv    = (const float*)d_in[6];
    const float* gamma = (const float*)d_in[7];
    float* out = (float*)d_out;

    char* ws = (char*)d_ws;
    bf16_t* W  = (bf16_t*)(ws + WS_W);
    bf16_t* Q  = (bf16_t*)(ws + WS_Q);
    bf16_t* Kp = (bf16_t*)(ws + WS_K);
    bf16_t* V  = (bf16_t*)(ws + WS_V);

    hipLaunchKernelGGL(wconv_kernel, dim3(OC_TOT), dim3(CC), 0, stream, wq, wk, wv, W);
    hipLaunchKernelGGL(qkv_proj_kernel, dim3(NN / 64, BB), dim3(128), 0, stream,
                       x, bq, bk, bv, W, Q, Kp, V);
    hipLaunchKernelGGL(attn_kernel, dim3(NN / 32, BB), dim3(128), 0, stream,
                       x, gamma, Q, Kp, V, out);
}
